// FocalTransformerBlock_58119497450122
// MI455X (gfx1250) — matrix operand order repaired and checked by reading, not compiled, not run on hardware
//
#include <hip/hip_runtime.h>
#include <hip/hip_bf16.h>

// ---------------------------------------------------------------------------
// Focal Transformer block for MI455X (gfx1250, wave32, WMMA f16 -> f32 acc)
// ---------------------------------------------------------------------------

typedef __attribute__((ext_vector_type(16))) _Float16 v16h;
typedef __attribute__((ext_vector_type(8)))  float    v8f;

#define B_   8
#define H_   112
#define W_   112
#define C_   256
#define NH_  8
#define HD_  32
#define WS_  7
#define WA_  49
#define NKR_ 132
#define NKALL_ 181
#define NKPAD_ 192
#define DFF_ 1024
#define NTOK_ (B_ * H_ * W_)          // 100352
#define NWIN_ 2048                     // 8 * 16 * 16
#define SCALE_ 0.17677669529663687f    // 1/sqrt(32)

// -------------------- CDNA5 async global->LDS copy path --------------------

#if defined(__has_builtin)
#if __has_builtin(__builtin_amdgcn_global_load_async_to_lds_b128)
#define HAVE_ASYNC_LDS 1
#endif
#endif
#ifndef HAVE_ASYNC_LDS
#define HAVE_ASYNC_LDS 0
#endif

// builtin signature wants 4xi32-vector pointees in AS(1)/AS(3)
typedef int v4i_t __attribute__((vector_size(16)));
typedef __attribute__((address_space(1))) v4i_t* gv4i_p;
typedef __attribute__((address_space(3))) v4i_t* lv4i_p;

// copy 16 bytes global -> LDS (async on gfx1250 toolchains that expose it)
__device__ __forceinline__ void copy16_g2l(const _Float16* gsrc,
                                           _Float16* ldst) {
#if HAVE_ASYNC_LDS
  __builtin_amdgcn_global_load_async_to_lds_b128(
      (gv4i_p)(v4i_t*)const_cast<_Float16*>(gsrc),
      (lv4i_p)(v4i_t*)ldst, 0, 0);
#else
  *(uint4*)ldst = *(const uint4*)gsrc;
#endif
}

__device__ __forceinline__ void wait_async_all() {
#if HAVE_ASYNC_LDS
#if __has_builtin(__builtin_amdgcn_s_wait_asynccnt)
  __builtin_amdgcn_s_wait_asynccnt(0);
#else
  asm volatile("s_wait_asynccnt 0x0" ::: "memory");
#endif
#endif
}

// ------------------------------- helpers -----------------------------------

__device__ __forceinline__ v8f wmma16(v16h a, v16h b, v8f c) {
  return __builtin_amdgcn_wmma_f32_16x16x32_f16(
      /*neg_a=*/false, a, /*neg_b=*/false, b,
      /*c_mod=*/(short)0, c, /*reuse_a=*/false, /*reuse_b=*/false);
}

// load 16 halves (8 from p0, 8 from p1) as two 16B vector loads
__device__ __forceinline__ v16h ld_frag2(const _Float16* p0, const _Float16* p1) {
  union { v16h v; uint4 q[2]; } u;
  u.q[0] = *(const uint4*)p0;
  u.q[1] = *(const uint4*)p1;
  return u.v;
}

__device__ __forceinline__ float gelu_exact(float x) {
  return 0.5f * x * (1.0f + erff(x * 0.7071067811865475f));
}

// ------------------------------- convert -----------------------------------

__global__ void cvt_f32_to_f16(const float* __restrict__ in,
                               _Float16* __restrict__ out, int n) {
  int i = blockIdx.x * blockDim.x + threadIdx.x;
  if (i < n) out[i] = (_Float16)in[i];
}

// ------------------------------- layernorm ---------------------------------

__global__ __launch_bounds__(C_) void ln_f32_to_f16(
    const float* __restrict__ x, const float* __restrict__ g,
    const float* __restrict__ b, _Float16* __restrict__ out) {
  __shared__ float s1[C_], s2[C_];
  const int t = blockIdx.x, c = threadIdx.x;
  float v = x[(size_t)t * C_ + c];
  s1[c] = v; s2[c] = v * v;
  __syncthreads();
  for (int off = C_ / 2; off > 0; off >>= 1) {
    if (c < off) { s1[c] += s1[c + off]; s2[c] += s2[c + off]; }
    __syncthreads();
  }
  const float mean = s1[0] * (1.0f / C_);
  const float var  = s2[0] * (1.0f / C_) - mean * mean;
  const float rstd = rsqrtf(var + 1e-5f);
  out[(size_t)t * C_ + c] = (_Float16)((v - mean) * rstd * g[c] + b[c]);
}

// ------------------------------- WMMA GEMM ---------------------------------
// C[M,N] = act(A[M,K] @ B[K,N] + bias) (+ resid).  M%64==0, N%128==0, K%32==0.
// Block tile 64x128, 8 waves: wm in 0..3 (16 rows), wn in 0..1 (64 cols ->
// 4 N-tiles, 4 WMMAs per wave per 32-K step).
// A tile staged via async global->LDS b128; B tile staged through registers
// with an in-flight transpose so both fragment loads are ds_load_b128.

template <int OUT_F16, int ACT_GELU, int RESID>
__global__ __launch_bounds__(256) void gemm_wmma(
    const _Float16* __restrict__ A, const _Float16* __restrict__ Bw,
    const float* __restrict__ bias, const float* __restrict__ resid,
    void* __restrict__ Out, int M, int N, int K) {
  __shared__ __align__(16) _Float16 As[64 * 32];     // [row][k]
  __shared__ __align__(16) _Float16 Bst[128 * 32];   // transposed: [col][k]
  const int tid  = threadIdx.x;
  const int lane = tid & 31, wid = tid >> 5;
  const int wm = wid & 3, wn = wid >> 2;
  const int bm = blockIdx.y, bn = blockIdx.x;
  const int lm = lane & 15, lh = lane >> 4;

  v8f acc[4] = {};

  // A staging: 64x32 halves = 256 threads x one 16B chunk
  const int baseA = tid * 8;
  const int rowA = baseA >> 5, kkA = baseA & 31;
  // B staging: 32x128 halves = 256 threads x two 16B chunks
  const int kkB = tid >> 3;             // 0..31
  const int colB = (tid & 7) * 16;      // 0..112 step 16

  for (int k0 = 0; k0 < K; k0 += 32) {
    // async copy A 64x32 -> LDS (row-major, layout preserved)
    copy16_g2l(A + (size_t)(bm * 64 + rowA) * K + k0 + kkA, As + baseA);
    // B 32x128 -> registers -> transposed LDS [col][k]
    {
      union { uint4 q; _Float16 h[8]; } u0, u1;
      const _Float16* bp = Bw + (size_t)(k0 + kkB) * N + bn * 128 + colB;
      u0.q = *(const uint4*)bp;
      u1.q = *(const uint4*)(bp + 8);
#pragma unroll
      for (int j = 0; j < 8; ++j) Bst[(colB + j) * 32 + kkB] = u0.h[j];
#pragma unroll
      for (int j = 0; j < 8; ++j) Bst[(colB + 8 + j) * 32 + kkB] = u1.h[j];
    }
    if (k0 + 32 < K) {
      __builtin_prefetch(Bw + (size_t)(k0 + 32 + kkB) * N + bn * 128 + colB, 0, 1);
    }
    wait_async_all();     // wave-local ASYNCcnt drain, then workgroup barrier
    __syncthreads();

    // A fragment: row = wm*16+lm, K split by lane half (ISA 16-bit A layout)
    const _Float16* ap = As + (wm * 16 + lm) * 32 + lh * 8;
    v16h af = ld_frag2(ap, ap + 16);
    // B fragments: col = lane&15, K split by lane half (same order as A)
#pragma unroll
    for (int t = 0; t < 4; ++t) {
      const _Float16* bp = Bst + (wn * 64 + t * 16 + lm) * 32 + lh * 8;
      v16h bf = ld_frag2(bp, bp + 16);
      acc[t] = wmma16(af, bf, acc[t]);
    }
    __syncthreads();
  }

  // epilogue: D layout: col = lane&15, row = lanehalf*8 + i
  const int r0 = bm * 64 + wm * 16 + lh * 8;
  const int c0 = bn * 128 + wn * 64 + lm;
#pragma unroll
  for (int i = 0; i < 8; ++i) {
    const int r = r0 + i;
#pragma unroll
    for (int t = 0; t < 4; ++t) {
      const int c = c0 + t * 16;
      float v = acc[t][i] + bias[c];
      if (ACT_GELU) v = gelu_exact(v);
      if (RESID) v += resid[(size_t)r * N + c];
      if (OUT_F16) ((_Float16*)Out)[(size_t)r * N + c] = (_Float16)v;
      else         ((float*)Out)[(size_t)r * N + c] = v;
    }
  }
}

// ------------------------------- attention ---------------------------------
// One block = one (window, head). 128 threads = 4 waves, one 16-query tile
// each. S = QK^T via WMMA (12 key tiles), register softmax (+bias, +mask),
// O = P V via WMMA (2 d-tiles x 6 k-steps).

__global__ __launch_bounds__(128) void attn_kernel(
    const _Float16* __restrict__ qkv,       // [NTOK][3C] f16 (q|k|v)
    const float* __restrict__ rpb_win,      // [169][8]
    const float* __restrict__ rpb_nb,       // [1][8][49][132]
    _Float16* __restrict__ outp) {          // [NTOK][C] f16
  __shared__ __align__(16) _Float16 Kall[NKPAD_ * HD_];   // [key][d]
  __shared__ __align__(16) _Float16 VallT[HD_ * NKPAD_];  // [d][key]
  __shared__ __align__(16) _Float16 Pbuf[64 * NKPAD_];    // [q][key]
  __shared__ short vmap[NKR_];

  const int tid  = threadIdx.x;
  const int head = blockIdx.x & 7;
  const int win  = blockIdx.x >> 3;
  const int b    = win >> 8;
  const int wi   = (win >> 4) & 15;
  const int wj   = win & 15;

  // rebuild VALID ring-token map (4 shifts x 33 valid of 49)
  if (tid == 0) {
    int c = 0;
    for (int s = 0; s < 4; ++s) {
      const int a = s >> 1, bb = s & 1;
      for (int t = 0; t < WA_; ++t) {
        const int ti = t / 7, tj = t % 7;
        const bool inR = a ? (ti >= 3) : (ti < 4);
        const bool inC = bb ? (tj >= 3) : (tj < 4);
        if (!(inR && inC)) vmap[c++] = (short)(s * WA_ + t);
      }
    }
  }
  __syncthreads();

  // stage K (row-major) and V (transposed) into LDS, zero-pad keys 181..191
  for (int idx = tid; idx < NKPAD_ * HD_; idx += 128) {
    const int key = idx >> 5, d = idx & 31;
    float kv = 0.f, vv = 0.f;
    if (key < NKALL_) {
      int hh, ww;
      if (key < WA_) {
        hh = wi * 7 + key / 7; ww = wj * 7 + key % 7;
      } else {
        const int g = vmap[key - WA_];
        const int s = g / WA_, t = g % WA_;
        const int dy = (s < 2) ? -3 : 3;
        const int dx = (s & 1) ? 3 : -3;
        hh = (wi * 7 + t / 7 - dy + H_) % H_;
        ww = (wj * 7 + t % 7 - dx + W_) % W_;
      }
      const size_t tok = (size_t)b * (H_ * W_) + (size_t)hh * W_ + ww;
      const _Float16* p = qkv + tok * (3 * C_) + head * HD_ + d;
      kv = (float)p[C_];          // k block
      vv = (float)p[2 * C_];      // v block
    }
    Kall[key * HD_ + d] = (_Float16)kv;
    VallT[d * NKPAD_ + key] = (_Float16)vv;
  }
  __syncthreads();

  const int wtile = tid >> 5;       // query tile 0..3
  const int lane  = tid & 31;
  const int lm = lane & 15, lh = lane >> 4;

  // Q fragment (vector load + f16 vector scale): row = wtile*16+lm (clamped)
  const int q0 = wtile * 16 + lm;
  const int qc = q0 < WA_ ? q0 : WA_ - 1;
  const size_t qtok =
      (size_t)b * (H_ * W_) + (size_t)(wi * 7 + qc / 7) * W_ + (wj * 7 + qc % 7);
  const _Float16* qp = qkv + qtok * (3 * C_) + head * HD_;
  v16h aq = ld_frag2(qp + lh * 8, qp + 16 + lh * 8);
  aq = aq * (_Float16)SCALE_;

  // S = Q K^T over 12 key tiles
  v8f S[12];
#pragma unroll
  for (int t = 0; t < 12; ++t) {
    const _Float16* kp = Kall + (t * 16 + lm) * HD_ + lh * 8;
    v16h bk = ld_frag2(kp, kp + 16);
    v8f c = {};
    S[t] = wmma16(aq, bk, c);
  }

  // bias + padding mask (D layout: col = lm, row = lh*8 + i)
#pragma unroll
  for (int t = 0; t < 12; ++t) {
    const int col = t * 16 + lm;
#pragma unroll
    for (int i = 0; i < 8; ++i) {
      const int qr = wtile * 16 + lh * 8 + i;
      float v = S[t][i];
      if (qr < WA_ && col < WA_) {
        const int rel = (qr / 7 - col / 7 + 6) * 13 + (qr % 7 - col % 7 + 6);
        v += rpb_win[rel * NH_ + head];
      } else if (qr < WA_ && col < NKALL_) {
        v += rpb_nb[((size_t)head * WA_ + qr) * NKR_ + (col - WA_)];
      } else {
        v = -1e30f;
      }
      S[t][i] = v;
    }
  }

  // register softmax: rows live per (lane-half, element); reduce across the
  // 16 lanes of each half with xor shuffles (masks 1,2,4,8 stay in-half)
  float mx[8], sm[8];
#pragma unroll
  for (int i = 0; i < 8; ++i) {
    float m = -1e30f;
#pragma unroll
    for (int t = 0; t < 12; ++t) m = fmaxf(m, S[t][i]);
    for (int off = 1; off < 16; off <<= 1) m = fmaxf(m, __shfl_xor(m, off, 32));
    mx[i] = m;
  }
#pragma unroll
  for (int i = 0; i < 8; ++i) {
    float s = 0.f;
#pragma unroll
    for (int t = 0; t < 12; ++t) {
      const float e = __expf(S[t][i] - mx[i]);
      S[t][i] = e; s += e;
    }
    for (int off = 1; off < 16; off <<= 1) s += __shfl_xor(s, off, 32);
    sm[i] = 1.0f / s;
  }
  // P -> LDS (f16), transposing from D layout to row-major [q][key]
#pragma unroll
  for (int t = 0; t < 12; ++t) {
#pragma unroll
    for (int i = 0; i < 8; ++i) {
      const int qr = wtile * 16 + lh * 8 + i;
      Pbuf[qr * NKPAD_ + t * 16 + lm] = (_Float16)(S[t][i] * sm[i]);
    }
  }
  __syncthreads();

  // O = P @ V : 2 d-tiles, K = 192 in 6 steps of 32
#pragma unroll
  for (int nt = 0; nt < 2; ++nt) {
    v8f o = {};
#pragma unroll
    for (int ks = 0; ks < 6; ++ks) {
      const _Float16* pp = Pbuf + (wtile * 16 + lm) * NKPAD_ + ks * 32 + lh * 8;
      v16h ap = ld_frag2(pp, pp + 16);
      const _Float16* vp = VallT + (nt * 16 + lm) * NKPAD_ + ks * 32 + lh * 8;
      v16h bv = ld_frag2(vp, vp + 16);
      o = wmma16(ap, bv, o);
    }
#pragma unroll
    for (int i = 0; i < 8; ++i) {
      const int qr = wtile * 16 + lh * 8 + i;
      if (qr < WA_) {
        const size_t tok = (size_t)b * (H_ * W_) +
                           (size_t)(wi * 7 + qr / 7) * W_ + (wj * 7 + qr % 7);
        outp[tok * C_ + head * HD_ + nt * 16 + lm] = (_Float16)o[i];
      }
    }
  }
}

// ------------------------------- launcher ----------------------------------

extern "C" void kernel_launch(void* const* d_in, const int* in_sizes, int n_in,
                              void* d_out, int out_size, void* d_ws,
                              size_t ws_size, hipStream_t stream) {
  const float* x       = (const float*)d_in[0];
  const float* g1      = (const float*)d_in[1];
  const float* b1n     = (const float*)d_in[2];
  const float* Wqkv    = (const float*)d_in[3];
  const float* bqkv    = (const float*)d_in[4];
  const float* rpb_win = (const float*)d_in[5];
  const float* rpb_nb  = (const float*)d_in[6];
  const float* Wp      = (const float*)d_in[7];
  const float* bp      = (const float*)d_in[8];
  const float* g2      = (const float*)d_in[9];
  const float* b2n     = (const float*)d_in[10];
  const float* W1      = (const float*)d_in[11];
  const float* b1m     = (const float*)d_in[12];
  const float* W2      = (const float*)d_in[13];
  const float* b2m     = (const float*)d_in[14];

  // workspace layout (bytes)
  constexpr size_t O_H16   = 0;                                    // 100352*256 f16
  constexpr size_t O_QKV16 = O_H16   + (size_t)NTOK_ * C_ * 2;     // 100352*768 f16
  constexpr size_t O_ATT16 = O_QKV16 + (size_t)NTOK_ * 3 * C_ * 2; // 100352*256 f16
  constexpr size_t O_X1    = O_ATT16 + (size_t)NTOK_ * C_ * 2;     // 100352*256 f32
  constexpr size_t O_FF16  = O_X1    + (size_t)NTOK_ * C_ * 4;     // 100352*1024 f16
  constexpr size_t O_W16   = O_FF16  + (size_t)NTOK_ * DFF_ * 2;   // weights f16

  char* ws = (char*)d_ws;
  _Float16* h16    = (_Float16*)(ws + O_H16);
  _Float16* qkv16  = (_Float16*)(ws + O_QKV16);
  _Float16* att16  = (_Float16*)(ws + O_ATT16);
  float*    x1     = (float*)   (ws + O_X1);
  _Float16* ff16   = (_Float16*)(ws + O_FF16);
  _Float16* w16    = (_Float16*)(ws + O_W16);
  _Float16* wqkv16 = w16;                       // 256*768
  _Float16* wp16   = w16 + 196608;              // 256*256
  _Float16* w1_16  = w16 + 262144;              // 256*1024
  _Float16* w2_16  = w16 + 524288;              // 1024*256

  // weights -> f16
  cvt_f32_to_f16<<<(196608 + 255) / 256, 256, 0, stream>>>(Wqkv, wqkv16, 196608);
  cvt_f32_to_f16<<<(65536  + 255) / 256, 256, 0, stream>>>(Wp,   wp16,   65536);
  cvt_f32_to_f16<<<(262144 + 255) / 256, 256, 0, stream>>>(W1,   w1_16,  262144);
  cvt_f32_to_f16<<<(262144 + 255) / 256, 256, 0, stream>>>(W2,   w2_16,  262144);

  // LN1
  ln_f32_to_f16<<<NTOK_, C_, 0, stream>>>(x, g1, b1n, h16);

  // QKV GEMM: (100352x256)@(256x768) -> f16
  gemm_wmma<1, 0, 0><<<dim3(768 / 128, NTOK_ / 64), 256, 0, stream>>>(
      h16, wqkv16, bqkv, nullptr, qkv16, NTOK_, 3 * C_, C_);

  // windowed focal attention
  attn_kernel<<<NWIN_ * NH_, 128, 0, stream>>>(qkv16, rpb_win, rpb_nb, att16);

  // proj GEMM + residual(x) -> x1 (f32)
  gemm_wmma<0, 0, 1><<<dim3(C_ / 128, NTOK_ / 64), 256, 0, stream>>>(
      att16, wp16, bp, x, x1, NTOK_, C_, C_);

  // LN2
  ln_f32_to_f16<<<NTOK_, C_, 0, stream>>>(x1, g2, b2n, h16);

  // MLP1 GEMM + GELU -> f16
  gemm_wmma<1, 1, 0><<<dim3(DFF_ / 128, NTOK_ / 64), 256, 0, stream>>>(
      h16, w1_16, b1m, nullptr, ff16, NTOK_, DFF_, C_);

  // MLP2 GEMM + residual(x1) -> d_out (f32)
  gemm_wmma<0, 0, 1><<<dim3(C_ / 128, NTOK_ / 64), 256, 0, stream>>>(
      ff16, w2_16, b2m, x1, (float*)d_out, NTOK_, C_, DFF_);
}
